// GPNN_87952340288006
// MI455X (gfx1250) — compile-verified, matrix operand order checked by reading, not run on hardware
//
#include <hip/hip_runtime.h>
#include <hip/hip_bf16.h>

// ---------------------------------------------------------------------------
// GPNN forward for MI455X (gfx1250): all large GEMMs via v_wmma_f32_16x16x32_f16
// with double-buffered async LDS staging (GLOBAL_LOAD_ASYNC_TO_LDS_B128).
// ---------------------------------------------------------------------------

typedef _Float16 h8   __attribute__((ext_vector_type(8)));
typedef _Float16 h4   __attribute__((ext_vector_type(4)));
typedef _Float16 v16h __attribute__((ext_vector_type(16)));
typedef float    v8f  __attribute__((ext_vector_type(8)));
typedef int      v4i  __attribute__((ext_vector_type(4)));

#define GEMM_RELU   1
#define GEMM_F16OUT 2

#if defined(__gfx1250__) && __has_builtin(__builtin_amdgcn_global_load_async_to_lds_b128)
#define HAVE_ASYNC 1
#else
#define HAVE_ASYNC 0
#endif

typedef __attribute__((address_space(1))) v4i gas_v4i;
typedef __attribute__((address_space(3))) v4i las_v4i;

__device__ __forceinline__ void stage16(const _Float16* g, _Float16* l) {
#if HAVE_ASYNC
  __builtin_amdgcn_global_load_async_to_lds_b128(
      (gas_v4i*)(void*)(const_cast<_Float16*>(g)),
      (las_v4i*)(void*)(l), 0, 0);
#else
  *(uint4*)l = *(const uint4*)g;
#endif
}

template <int N>
__device__ __forceinline__ void async_wait() {
#if HAVE_ASYNC
#if __has_builtin(__builtin_amdgcn_s_wait_asynccnt)
  __builtin_amdgcn_s_wait_asynccnt(N);
#else
  asm volatile("s_wait_asynccnt %0" ::"n"(N) : "memory");
#endif
#endif
}

__device__ __forceinline__ v8f wmma16x16x32(v16h a, v16h b, v8f c) {
  return __builtin_amdgcn_wmma_f32_16x16x32_f16(false, a, false, b, (short)0, c,
                                                false, false);
}

// ---------------------------------------------------------------------------
// Tiled WMMA GEMM:  D[M,N] = act(A[M,K] @ W[N,K]^T + bias)
// A, W f16 row-major.  D f32 or f16 (flags).  M%128==0, N%128==0, K%32==0.
// Block = 256 threads (8 waves).  Block tile 128x128.
// Wave tile 32x64 = 2x4 WMMA (16x16x32) patches, f32 accumulate.
// Double-buffered LDS staged with async global->LDS copies.
// ---------------------------------------------------------------------------
#define BM 128
#define BN 128
#define BK 32
#define LDS_STRIDE 40  // halves; multiple of 8 for 16B-aligned fragment loads
#define TILE_HALVES (128 * LDS_STRIDE)
#define STAGE_OPS 4  // async b128 ops issued per thread per tile (A:2 + B:2)

__global__ __launch_bounds__(256) void gemm_f16_wmma(
    const _Float16* __restrict__ A, const _Float16* __restrict__ W,
    const float* __restrict__ bias, void* __restrict__ D,
    int M, int N, int K, int flags) {
  __shared__ __align__(16) _Float16 sA[2 * TILE_HALVES];
  __shared__ __align__(16) _Float16 sB[2 * TILE_HALVES];

  const int tid  = threadIdx.x;
  const int lane = tid & 31;
  const int wave = tid >> 5;
  const int m0 = blockIdx.x * BM;
  const int n0 = blockIdx.y * BN;
  const int wm = (wave & 3) * 32;   // wave row offset inside block tile
  const int wn = (wave >> 2) * 64;  // wave col offset inside block tile

  const int col16 = lane & 15;
  const int khalf = (lane >> 4) * 8;  // lanes 0-15: K 0..7/16..23 ; 16-31: 8..15/24..31

  // Per-thread staging coordinates: 512 x uint4 per tile each for A and B.
  const int sr = tid >> 2;         // rows tid/4 and tid/4 + 64
  const int sc = (tid & 3) * 8;    // half offset within 32-half row

  v8f acc[2][4] = {};

  const int nk = K / BK;
  // Prologue: stage tile 0 into buffer 0.
  {
    const size_t gk = 0;
    stage16(&A[(size_t)(m0 + sr) * K + gk + sc], &sA[sr * LDS_STRIDE + sc]);
    stage16(&A[(size_t)(m0 + sr + 64) * K + gk + sc],
            &sA[(sr + 64) * LDS_STRIDE + sc]);
    stage16(&W[(size_t)(n0 + sr) * K + gk + sc], &sB[sr * LDS_STRIDE + sc]);
    stage16(&W[(size_t)(n0 + sr + 64) * K + gk + sc],
            &sB[(sr + 64) * LDS_STRIDE + sc]);
  }

  for (int kt = 0; kt < nk; ++kt) {
    const int cur = kt & 1;
    if (kt + 1 < nk) {
      // Stage next tile into the other buffer while current tile computes.
      const int nxt = cur ^ 1;
      const size_t gk = (size_t)(kt + 1) * BK;
      _Float16* dA = &sA[nxt * TILE_HALVES];
      _Float16* dB = &sB[nxt * TILE_HALVES];
      stage16(&A[(size_t)(m0 + sr) * K + gk + sc], &dA[sr * LDS_STRIDE + sc]);
      stage16(&A[(size_t)(m0 + sr + 64) * K + gk + sc],
              &dA[(sr + 64) * LDS_STRIDE + sc]);
      stage16(&W[(size_t)(n0 + sr) * K + gk + sc], &dB[sr * LDS_STRIDE + sc]);
      stage16(&W[(size_t)(n0 + sr + 64) * K + gk + sc],
              &dB[(sr + 64) * LDS_STRIDE + sc]);
      async_wait<STAGE_OPS>();  // retire tile kt; tile kt+1 stays in flight
    } else {
      async_wait<0>();
    }
    __syncthreads();

    const _Float16* bufA = &sA[cur * TILE_HALVES];
    const _Float16* bufB = &sB[cur * TILE_HALVES];
    v16h afrag[2], bfrag[4];
    #pragma unroll
    for (int i = 0; i < 2; ++i) {
      const _Float16* p = &bufA[(wm + i * 16 + col16) * LDS_STRIDE + khalf];
      h8 lo = *(const h8*)p;
      h8 hi = *(const h8*)(p + 16);
      afrag[i] = __builtin_shufflevector(lo, hi, 0, 1, 2, 3, 4, 5, 6, 7, 8, 9,
                                         10, 11, 12, 13, 14, 15);
    }
    #pragma unroll
    for (int j = 0; j < 4; ++j) {
      const _Float16* p = &bufB[(wn + j * 16 + col16) * LDS_STRIDE + khalf];
      h8 lo = *(const h8*)p;
      h8 hi = *(const h8*)(p + 16);
      bfrag[j] = __builtin_shufflevector(lo, hi, 0, 1, 2, 3, 4, 5, 6, 7, 8, 9,
                                         10, 11, 12, 13, 14, 15);
    }
    #pragma unroll
    for (int i = 0; i < 2; ++i)
      #pragma unroll
      for (int j = 0; j < 4; ++j)
        acc[i][j] = wmma16x16x32(afrag[i], bfrag[j], acc[i][j]);
    __syncthreads();
  }

  // Epilogue: lane holds col n = lane&15, rows (lane>=16?8:0)+r of each 16x16 tile
  const bool relu   = flags & GEMM_RELU;
  const bool f16out = flags & GEMM_F16OUT;
  const int mbase = (lane >> 4) * 8;
  #pragma unroll
  for (int i = 0; i < 2; ++i) {
    #pragma unroll
    for (int j = 0; j < 4; ++j) {
      const int gn = n0 + wn + j * 16 + col16;
      const float bv = bias ? bias[gn] : 0.f;
      #pragma unroll
      for (int r = 0; r < 8; ++r) {
        const int gm = m0 + wm + i * 16 + mbase + r;
        float v = acc[i][j][r] + bv;
        if (relu) v = fmaxf(v, 0.f);
        if (f16out)
          ((_Float16*)D)[(size_t)gm * N + gn] = (_Float16)v;
        else
          ((float*)D)[(size_t)gm * N + gn] = v;
      }
    }
  }
}

// ---------------------------------------------------------------------------
// f32 -> f16 conversion (n must be a multiple of 4)
// ---------------------------------------------------------------------------
__global__ void cvt_f32_f16(const float* __restrict__ s, _Float16* __restrict__ d,
                            int n4) {
  int i = blockIdx.x * 256 + threadIdx.x;
  if (i < n4) {
    float4 v = ((const float4*)s)[i];
    h4 o = {(_Float16)v.x, (_Float16)v.y, (_Float16)v.z, (_Float16)v.w};
    ((h4*)d)[i] = o;
  }
}

__global__ void copy_f32(const float* __restrict__ s, float* __restrict__ d, int n) {
  int i = blockIdx.x * 256 + threadIdx.x;
  if (i < n) d[i] = s[i];
}

// ---------------------------------------------------------------------------
// pred_adj[row] = mask ? a2[row,:] . w3 + b3 : 0     (one wave per row)
// ---------------------------------------------------------------------------
__global__ void link3_pred_kernel(const _Float16* __restrict__ a2,
                                  const float* __restrict__ w3,
                                  const float* __restrict__ b3,
                                  const int* __restrict__ num_obj,
                                  float* __restrict__ pred, int N, int Kd) {
  int row  = blockIdx.x * 8 + (threadIdx.x >> 5);
  int lane = threadIdx.x & 31;
  float s = 0.f;
  const _Float16* ap = a2 + (size_t)row * Kd;
  for (int k = lane; k < Kd; k += 32) s += (float)ap[k] * w3[k];
  #pragma unroll
  for (int off = 16; off > 0; off >>= 1) s += __shfl_xor(s, off, 32);
  if (lane == 0) {
    int b = row / (N * N);
    int rem = row % (N * N);
    int v = rem / N, w = rem % N;
    int no = num_obj[b];
    pred[row] = (v < no && w < no) ? (s + b3[0]) : 0.f;
  }
}

// ---------------------------------------------------------------------------
// Fused message / aggregation / edge-state transpose.
// Block = (b,v), thread = feature m.  Loops over w.
//   msg = sigmoid(pred[b,v,w]) * relu(hn[b,w]+en[b,v,w]+bm)
//   m_sum[b,v] = sum_{w valid} msg
//   E_new[b,w,v] = mask2 ? msg : E_old[b,w,v]
// ---------------------------------------------------------------------------
__global__ void msg_kernel(const float* __restrict__ pred,
                           const _Float16* __restrict__ hn,
                           const _Float16* __restrict__ en,
                           const float* __restrict__ bm_p,
                           const _Float16* __restrict__ E_old,
                           _Float16* __restrict__ E_new,
                           float* __restrict__ m_sum,
                           const int* __restrict__ num_obj, int N, int M) {
  const int b = blockIdx.x / N;
  const int v = blockIdx.x % N;
  const int m = threadIdx.x;
  const int no = num_obj[b];
  const bool vvalid = v < no;
  const float bias = bm_p[m];
  float acc = 0.f;
  const size_t row_bv = (size_t)(b * N + v);
  for (int w = 0; w < N; ++w) {
    float g = 1.f / (1.f + __expf(-pred[row_bv * N + w]));
    float val = (float)hn[(size_t)(b * N + w) * M + m] +
                (float)en[(row_bv * N + w) * M + m] + bias;
    float ms = g * fmaxf(val, 0.f);
    bool wvalid = w < no;
    if (wvalid) acc += ms;
    size_t eidx = (((size_t)(b * N + w)) * N + v) * M + m;
    E_new[eidx] = (vvalid && wvalid) ? (_Float16)ms : E_old[eidx];
  }
  m_sum[row_bv * M + m] = acc;
}

// ---------------------------------------------------------------------------
// GRU update (block = node row, thread = feature)
// ---------------------------------------------------------------------------
__global__ void gru_kernel(const float* __restrict__ gi,
                           const float* __restrict__ gh,
                           const int* __restrict__ num_obj,
                           float* __restrict__ H, int N, int M) {
  int row = blockIdx.x;
  int m = threadIdx.x;
  int b = row / N, v = row % N;
  if (v >= num_obj[b]) return;
  const float* gir = gi + (size_t)row * 3 * M;
  const float* ghr = gh + (size_t)row * 3 * M;
  float r = 1.f / (1.f + __expf(-(gir[m] + ghr[m])));
  float z = 1.f / (1.f + __expf(-(gir[M + m] + ghr[M + m])));
  float n = tanhf(gir[2 * M + m] + r * ghr[2 * M + m]);
  size_t hi = (size_t)row * M + m;
  float h = H[hi];
  H[hi] = (1.f - z) * n + z * h;
}

// ---------------------------------------------------------------------------
// ci[b,p] = 0.5*(H[b,i0] + H[b,i1])  (output f16, consumed by GEMM)
// ---------------------------------------------------------------------------
__global__ void gather_pairs(const float* __restrict__ H,
                             const int* __restrict__ pairs,
                             _Float16* __restrict__ ci, int N, int P, int M) {
  int idx = blockIdx.x * 256 + threadIdx.x;  // over B*P*M
  int m = idx & (M - 1);
  int p = idx >> 8;  // M == 256
  int b = p / P;
  int i0 = pairs[p * 2 + 0];
  int i1 = pairs[p * 2 + 1];
  float v = 0.5f * (H[((size_t)(b * N + i0)) * M + m] +
                    H[((size_t)(b * N + i1)) * M + m]);
  ci[idx] = (_Float16)v;
}

// ---------------------------------------------------------------------------
// Tiny final classifier layer: out[r,o] = hid[r,:] . W2[o,:] + b2[o]
// ---------------------------------------------------------------------------
__global__ void mlp_out_kernel(const _Float16* __restrict__ hid,
                               const float* __restrict__ W2,
                               const float* __restrict__ b2,
                               float* __restrict__ out, int rows, int Kd,
                               int Nout) {
  int idx = blockIdx.x * 256 + threadIdx.x;
  if (idx >= rows * Nout) return;
  int r = idx / Nout, o = idx % Nout;
  const _Float16* hp = hid + (size_t)r * Kd;
  const float* wp = W2 + (size_t)o * Kd;
  float s = b2[o];
  for (int k = 0; k < Kd; ++k) s += (float)hp[k] * wp[k];
  out[idx] = s;
}

// ---------------------------------------------------------------------------
// Host orchestration
// ---------------------------------------------------------------------------
static inline void run_gemm(const _Float16* A, const _Float16* W,
                            const float* bias, void* D, int M, int N, int K,
                            int flags, hipStream_t s) {
  dim3 grid(M / BM, N / BN);
  gemm_f16_wmma<<<grid, 256, 0, s>>>(A, W, bias, D, M, N, K, flags);
}

static inline void run_cvt(const float* s, _Float16* d, int n, hipStream_t st) {
  int n4 = n / 4;
  cvt_f32_f16<<<(n4 + 255) / 256, 256, 0, st>>>(s, d, n4);
}

extern "C" void kernel_launch(void* const* d_in, const int* in_sizes, int n_in,
                              void* d_out, int out_size, void* d_ws,
                              size_t ws_size, hipStream_t stream) {
  const int B = 16, N = 64, P = 128, NF = 1024, EF = 256, M = 256, H = 512;
  const int NN = B * N * N;      // 65536 edge rows
  const int NR = B * N;          // 1024 node rows
  const int PR = B * P;          // 2048 pair rows

  // ---- inputs (setup_inputs dict order) ----
  const float* rsf  = (const float*)d_in[0];   // [B,N,N,EF]
  const float* cnf  = (const float*)d_in[1];   // [B,N,NF]
  /* d_in[2] activity_embedding: unused by reference */
  const int* num_obj = (const int*)d_in[3];
  const int* pairs   = (const int*)d_in[4];
  const float* W_edge_rs = (const float*)d_in[5];
  const float* b_edge_rs = (const float*)d_in[6];
  const float* W_node_rs = (const float*)d_in[7];
  const float* b_node_rs = (const float*)d_in[8];
  const float* link_W1 = (const float*)d_in[9];
  const float* link_b1 = (const float*)d_in[10];
  const float* link_W2 = (const float*)d_in[11];
  const float* link_b2 = (const float*)d_in[12];
  const float* link_W3 = (const float*)d_in[13];
  const float* link_b3 = (const float*)d_in[14];
  const float* Wm_node = (const float*)d_in[15];
  const float* Wm_edge = (const float*)d_in[16];
  const float* bm_p    = (const float*)d_in[17];
  const float* Wih = (const float*)d_in[18];
  const float* Whh = (const float*)d_in[19];
  const float* bih = (const float*)d_in[20];
  const float* bhh = (const float*)d_in[21];
  const float* lr_W1 = (const float*)d_in[22];
  const float* lr_b1 = (const float*)d_in[23];
  const float* lr_W2 = (const float*)d_in[24];
  const float* lr_b2 = (const float*)d_in[25];
  const float* cr_W1 = (const float*)d_in[26];
  const float* cr_b1 = (const float*)d_in[27];
  const float* cr_W2 = (const float*)d_in[28];
  const float* cr_b2 = (const float*)d_in[29];
  const float* mr_W1 = (const float*)d_in[30];
  const float* mr_b1 = (const float*)d_in[31];
  const float* mr_W2 = (const float*)d_in[32];
  const float* mr_b2 = (const float*)d_in[33];

  // ---- workspace layout ----
  size_t off = 0;
  auto alloc = [&](size_t bytes) -> char* {
    char* p = (char*)d_ws + off;
    off += (bytes + 255) & ~(size_t)255;
    return p;
  };
  _Float16* Ebuf0 = (_Float16*)alloc((size_t)NN * M * 2);   // E0 / E ping
  _Float16* Ebuf1 = (_Float16*)alloc((size_t)NN * M * 2);   // E pong
  _Float16* en_h  = (_Float16*)alloc((size_t)NN * M * 2);
  _Float16* a1_h  = (_Float16*)alloc((size_t)NN * H * 2);
  _Float16* a2_h  = (_Float16*)alloc((size_t)NN * H * 2);
  // aliases: conversion staging lives in a1/a2 (dead before a1/a2 are written)
  _Float16* rsf_h = a1_h;  // [NN, EF] f16  (32 MiB < 64 MiB)
  _Float16* cnf_h = a2_h;  // [NR, NF] f16  ( 2 MiB)
  float*    predb = (float*)alloc((size_t)NN * 4);
  float*    Hbuf  = (float*)alloc((size_t)NR * M * 4);
  _Float16* Hh    = (_Float16*)alloc((size_t)NR * M * 2);
  _Float16* hn_h  = (_Float16*)alloc((size_t)NR * M * 2);
  float*    m_sum = (float*)alloc((size_t)NR * M * 4);
  _Float16* m_h   = (_Float16*)alloc((size_t)NR * M * 2);
  float*    gi    = (float*)alloc((size_t)NR * 3 * M * 4);
  float*    gh    = (float*)alloc((size_t)NR * 3 * M * 4);
  _Float16* ci_h  = (_Float16*)alloc((size_t)PR * M * 2);
  _Float16* hid_h = (_Float16*)alloc((size_t)PR * H * 2);
  _Float16* Wer_h = (_Float16*)alloc((size_t)M * EF * 2);
  _Float16* Wnr_h = (_Float16*)alloc((size_t)M * NF * 2);
  _Float16* lW1_h = (_Float16*)alloc((size_t)H * M * 2);
  _Float16* lW2_h = (_Float16*)alloc((size_t)H * H * 2);
  _Float16* Wmn_h = (_Float16*)alloc((size_t)M * M * 2);
  _Float16* Wme_h = (_Float16*)alloc((size_t)M * M * 2);
  _Float16* Wih_h = (_Float16*)alloc((size_t)3 * M * M * 2);
  _Float16* Whh_h = (_Float16*)alloc((size_t)3 * M * M * 2);
  _Float16* lrW_h = (_Float16*)alloc((size_t)H * M * 2);
  _Float16* crW_h = (_Float16*)alloc((size_t)H * M * 2);
  _Float16* mrW_h = (_Float16*)alloc((size_t)H * M * 2);

  // ---- conversions to f16 ----
  run_cvt(rsf, rsf_h, NN * EF, stream);
  run_cvt(cnf, cnf_h, NR * NF, stream);
  run_cvt(W_edge_rs, Wer_h, M * EF, stream);
  run_cvt(W_node_rs, Wnr_h, M * NF, stream);
  run_cvt(link_W1, lW1_h, H * M, stream);
  run_cvt(link_W2, lW2_h, H * H, stream);
  run_cvt(Wm_node, Wmn_h, M * M, stream);
  run_cvt(Wm_edge, Wme_h, M * M, stream);
  run_cvt(Wih, Wih_h, 3 * M * M, stream);
  run_cvt(Whh, Whh_h, 3 * M * M, stream);
  run_cvt(lr_W1, lrW_h, H * M, stream);
  run_cvt(cr_W1, crW_h, H * M, stream);
  run_cvt(mr_W1, mrW_h, H * M, stream);

  // ---- feature resize ----
  run_gemm(rsf_h, Wer_h, b_edge_rs, Ebuf0, NN, M, EF, GEMM_F16OUT, stream); // E0
  run_gemm(cnf_h, Wnr_h, b_node_rs, Hbuf, NR, M, NF, 0, stream);            // H (f32)
  run_gemm(Ebuf0, Wme_h, nullptr, en_h, NN, M, M, GEMM_F16OUT, stream);     // en

  _Float16* Ecur = Ebuf0;
  _Float16* Enxt = Ebuf1;
  for (int layer = 0; layer < 3; ++layer) {
    // link MLP over edges
    run_gemm(Ecur, lW1_h, link_b1, a1_h, NN, H, M, GEMM_RELU | GEMM_F16OUT, stream);
    run_gemm(a1_h, lW2_h, link_b2, a2_h, NN, H, H, GEMM_RELU | GEMM_F16OUT, stream);
    link3_pred_kernel<<<NN / 8, 256, 0, stream>>>(a2_h, link_W3, link_b3,
                                                  num_obj, predb, N, H);
    // node-side message linear
    run_cvt(Hbuf, Hh, NR * M, stream);
    run_gemm(Hh, Wmn_h, nullptr, hn_h, NR, M, M, GEMM_F16OUT, stream);
    // fused message / aggregate / edge-state transpose
    msg_kernel<<<NR, M, 0, stream>>>(predb, hn_h, en_h, bm_p, Ecur, Enxt,
                                     m_sum, num_obj, N, M);
    // GRU
    run_cvt(m_sum, m_h, NR * M, stream);
    run_gemm(m_h, Wih_h, bih, gi, NR, 3 * M, M, 0, stream);
    run_gemm(Hh, Whh_h, bhh, gh, NR, 3 * M, M, 0, stream);
    gru_kernel<<<NR, M, 0, stream>>>(gi, gh, num_obj, Hbuf, N, M);
    // swap edge state
    _Float16* t = Ecur; Ecur = Enxt; Enxt = t;
  }

  // ---- classifiers ----
  float* outp = (float*)d_out;
  gather_pairs<<<(PR * M) / 256, 256, 0, stream>>>(Hbuf, pairs, ci_h, N, P, M);

  run_gemm(ci_h, lrW_h, lr_b1, hid_h, PR, H, M, GEMM_RELU | GEMM_F16OUT, stream);
  mlp_out_kernel<<<(PR * 4 + 255) / 256, 256, 0, stream>>>(hid_h, lr_W2, lr_b2,
                                                           outp, PR, H, 4);
  run_gemm(ci_h, crW_h, cr_b1, hid_h, PR, H, M, GEMM_RELU | GEMM_F16OUT, stream);
  mlp_out_kernel<<<(PR * 6 + 255) / 256, 256, 0, stream>>>(hid_h, cr_W2, cr_b2,
                                                           outp + PR * 4, PR, H, 6);
  run_gemm(ci_h, mrW_h, mr_b1, hid_h, PR, H, M, GEMM_RELU | GEMM_F16OUT, stream);
  mlp_out_kernel<<<(PR * 17 + 255) / 256, 256, 0, stream>>>(hid_h, mr_W2, mr_b2,
                                                            outp + PR * 10, PR, H, 17);
  // pred_adj (from last layer) -> output
  copy_f32<<<NN / 256, 256, 0, stream>>>(predb, outp + PR * 27, NN);
  (void)in_sizes; (void)n_in; (void)out_size; (void)ws_size;
}
